// MixerModel_13546326851969
// MI455X (gfx1250) — compile-verified, matrix operand order checked by reading, not run on hardware
//
#include <hip/hip_runtime.h>

// MI455X (gfx1250) Mamba mixer. bf16 WMMA (f32 accum) for in_proj/out_proj
// (16Mx64N wave tiles, no EXEC divergence in the K loop), f32 register-resident
// selective scan with async global->LDS broadcast of B/C, fused elementwise
// kernels for embed / LN+residual / conv+SiLU / gate.

typedef __attribute__((ext_vector_type(8)))  __bf16 v8bf;
typedef __attribute__((ext_vector_type(16))) __bf16 v16bf;
typedef __attribute__((ext_vector_type(8)))  float  v8f;

namespace {
constexpr int kB   = 4;
constexpr int kL   = 1024;
constexpr int kT   = 2048;        // L + L padded rows
constexpr int kM   = kB * kT;     // 8192 GEMM rows
constexpr int kD   = 272;         // model width
constexpr int kDP  = 288;        // K padded to multiple of 32
constexpr int kDI  = 544;         // d_inner
constexpr int kNX  = 2 * kDI;     // 1088
constexpr int kDBL = 49;          // dt_rank + 2*d_state
constexpr int kDTR = 17;
constexpr int kNS  = 16;          // d_state
}

// ---------------------------------------------------------------- embed -----
__global__ void embed_kernel(const int* __restrict__ types,
                             const float* __restrict__ times,
                             const float* __restrict__ temp,
                             const float* __restrict__ emb,
                             float* __restrict__ h0) {
  int idx = blockIdx.x * blockDim.x + threadIdx.x;
  if (idx >= kM * kD) return;
  int c = idx % kD;
  int m = idx / kD;
  int b = m / kT;
  int t = m % kT;
  float v;
  if (t < kL) {
    if (c < 256) {
      v = tanhf(emb[(size_t)types[b * kL + t] * 256 + c]);
    } else {
      int j2 = ((c - 256) >> 1) * 2;                          // 0,2,...,14
      float div = __expf(-0.57564627324851142f * (float)j2);  // -(ln 1e4)/16
      float arg = times[b * kL + t] * div;
      v = ((c - 256) & 1) ? __cosf(arg) : __sinf(arg);
    }
  } else {
    v = (c < 256) ? 0.0f
                  : temp[((size_t)b * kL + (t - kL)) * 16 + (c - 256)];
  }
  h0[idx] = v;
}

// --------------------------------------------- f32 -> bf16 pad converter ----
__global__ void cvt_pad_kernel(const float* __restrict__ src,
                               __bf16* __restrict__ dst,
                               int rows, int K, int Kp) {
  int idx = blockIdx.x * blockDim.x + threadIdx.x;
  if (idx >= rows * Kp) return;
  int r = idx / Kp, c = idx % Kp;
  dst[idx] = (__bf16)((c < K) ? src[(size_t)r * K + c] : 0.0f);
}

// -------------------------------------------- residual add + layernorm ------
__global__ void __launch_bounds__(256)
addnorm_kernel(const float* __restrict__ hin, float* __restrict__ resid,
               const float* __restrict__ gw, const float* __restrict__ gb,
               __bf16* __restrict__ outb, int addFlag) {
  __shared__ float s1[256];
  __shared__ float s2[256];
  int m = blockIdx.x;
  int tid = threadIdx.x;
  size_t base = (size_t)m * kD;
  int c0 = tid, c1 = tid + 256;
  float a0 = hin[base + c0];
  if (addFlag) a0 += resid[base + c0];
  resid[base + c0] = a0;
  float a1 = 0.0f;
  bool has1 = (c1 < kD);
  if (has1) {
    a1 = hin[base + c1];
    if (addFlag) a1 += resid[base + c1];
    resid[base + c1] = a1;
  }
  s1[tid] = a0 + (has1 ? a1 : 0.0f);
  s2[tid] = a0 * a0 + (has1 ? a1 * a1 : 0.0f);
  __syncthreads();
  for (int o = 128; o > 0; o >>= 1) {
    if (tid < o) { s1[tid] += s1[tid + o]; s2[tid] += s2[tid + o]; }
    __syncthreads();
  }
  float mean = s1[0] * (1.0f / kD);
  float var  = s2[0] * (1.0f / kD) - mean * mean;
  float inv  = rsqrtf(var + 1e-5f);
  size_t ob = (size_t)m * kDP;
  outb[ob + c0] = (__bf16)((a0 - mean) * inv * gw[c0] + gb[c0]);
  if (has1) outb[ob + c1] = (__bf16)((a1 - mean) * inv * gw[c1] + gb[c1]);
  if (tid < kDP - kD) outb[ob + kD + tid] = (__bf16)0.0f;   // zero K pad
}

// --------------------------------------------------------- WMMA GEMM --------
// C[M,N] = A[M,Kp] * W[N,Kp]^T ; A,W bf16 row-major (ld = Kp), C f32 (ld=ldc).
// 16-bit A/B fragment layout per CDNA5 ISA 7.12.2: lane holds row (lane&15),
// K chunks [k+8*(lane>>4) .. +7] and [k+16+8*(lane>>4) .. +7].
__device__ __forceinline__ v16bf load_frag(const __bf16* __restrict__ base,
                                           int row, int ld, int k0, int lane) {
  const __bf16* p = base + (size_t)row * ld + k0 + ((lane >> 4) << 3);
  v8bf c0 = *reinterpret_cast<const v8bf*>(p);
  v8bf c1 = *reinterpret_cast<const v8bf*>(p + 16);
  v16bf r;
#pragma unroll
  for (int i = 0; i < 8; ++i) { r[i] = c0[i]; r[i + 8] = c1[i]; }
  return r;
}

// 8 waves stacked in M (128 rows) x 64 cols per block; each wave owns a
// 16x64 strip: one A fragment feeds 4 WMMAs per K step. Out-of-range N tiles
// read clamped (in-bounds) weight rows and are dropped at the store -- the
// K loop is branch-free, EXEC stays all-ones around every v_wmma.
__global__ void __launch_bounds__(256)
wmma_gemm_bf16(const __bf16* __restrict__ A, const __bf16* __restrict__ W,
               float* __restrict__ C, int M, int N, int Kp, int ldc) {
  const int lane  = threadIdx.x & 31;
  const int wave  = threadIdx.x >> 5;
  const int m0    = blockIdx.y * 128 + wave * 16;
  const int nbase = blockIdx.x * 64;
  const int arow  = m0 + (lane & 15);
  int wrow[4];
#pragma unroll
  for (int j = 0; j < 4; ++j) {
    int n0 = nbase + j * 16;
    wrow[j] = ((n0 < N) ? n0 : (N - 16)) + (lane & 15);   // clamped, in-bounds
  }
  v8f acc[4];
#pragma unroll
  for (int j = 0; j < 4; ++j)
#pragma unroll
    for (int i = 0; i < 8; ++i) acc[j][i] = 0.0f;

  for (int k = 0; k < Kp; k += 32) {
    if (k + 32 < Kp)  // hint next K tile toward L2/L0
      __builtin_prefetch(A + (size_t)arow * Kp + k + 32, 0, 1);
    v16bf a = load_frag(A, arow, Kp, k, lane);
#pragma unroll
    for (int j = 0; j < 4; ++j) {
      v16bf b = load_frag(W, wrow[j], Kp, k, lane);
      acc[j] = __builtin_amdgcn_wmma_f32_16x16x32_bf16(
          false, a, false, b, (short)0, acc[j], false, false);
    }
  }
#pragma unroll
  for (int j = 0; j < 4; ++j) {
    int n0 = nbase + j * 16;
    if (n0 < N) {
#pragma unroll
      for (int r = 0; r < 8; ++r) {
        int row = m0 + r + ((lane >> 4) << 3);
        C[(size_t)row * ldc + n0 + (lane & 15)] = acc[j][r];
      }
    }
  }
}

// ------------------------------------- depthwise causal conv (k=4) + SiLU ---
__global__ void conv_silu_kernel(const float* __restrict__ xz,
                                 const float* __restrict__ cw,
                                 const float* __restrict__ cb,
                                 float* __restrict__ xm,
                                 __bf16* __restrict__ xmb) {
  int idx = blockIdx.x * blockDim.x + threadIdx.x;
  if (idx >= kM * kDI) return;
  int d = idx % kDI;
  int m = idx / kDI;
  int t = m % kT;
  float acc = cb[d];
#pragma unroll
  for (int k = 0; k < 4; ++k) {
    int tt = t - 3 + k;
    if (tt >= 0)
      acc += cw[d * 4 + k] * xz[(size_t)(m - 3 + k) * kNX + d];
  }
  float s = acc / (1.0f + __expf(-acc));    // SiLU
  xm[idx]  = s;
  xmb[idx] = (__bf16)s;
}

// --------------------------------------- x_proj: dbl = xm @ xproj_w^T -------
__global__ void __launch_bounds__(64)
xproj_kernel(const float* __restrict__ xm, const float* __restrict__ xw,
             float* __restrict__ dbl) {
  __shared__ float row[kDI];
  int m = blockIdx.x;
  for (int c = threadIdx.x; c < kDI; c += 64) row[c] = xm[(size_t)m * kDI + c];
  __syncthreads();
  int j = threadIdx.x;
  if (j < kDBL) {
    const float* w = xw + (size_t)j * kDI;
    float acc = 0.0f;
    for (int c = 0; c < kDI; ++c) acc += row[c] * w[c];
    dbl[(size_t)m * kDBL + j] = acc;
  }
}

// ------------------------------- delta = softplus(dt @ dt_w^T + dt_b) -------
__global__ void delta_kernel(const float* __restrict__ dbl,
                             const float* __restrict__ dtw,
                             const float* __restrict__ dtb,
                             float* __restrict__ delta) {
  int idx = blockIdx.x * blockDim.x + threadIdx.x;
  if (idx >= kM * kDI) return;
  int d = idx % kDI;
  int m = idx / kDI;
  float acc = dtb[d];
#pragma unroll
  for (int r = 0; r < kDTR; ++r)
    acc += dbl[(size_t)m * kDBL + r] * dtw[d * kDTR + r];
  delta[idx] = (acc > 20.0f) ? acc : log1pf(__expf(acc));
}

// --------------------------------------------------- selective scan ---------
// one thread per (b, d) channel, 16 states in registers. The per-timestep
// (B|C) 32-float vector is broadcast to all 17 waves via CDNA5 async
// global->LDS copy (GLOBAL_LOAD_ASYNC_TO_LDS_B32, tracked by ASYNCcnt),
// issued by wave 0 with full EXEC.
__global__ void __launch_bounds__(kDI)
scan_kernel(const float* __restrict__ xm, const float* __restrict__ delta,
            const float* __restrict__ dbl, const float* __restrict__ A_log,
            const float* __restrict__ Dp, float* __restrict__ y) {
  __shared__ float BC[2 * kNS];
  int b = blockIdx.x, d = threadIdx.x;
  float A[kNS], h[kNS];
  float dparam = Dp[d];
#pragma unroll
  for (int n = 0; n < kNS; ++n) {
    A[n] = -__expf(A_log[d * kNS + n]);
    h[n] = 0.0f;
  }
  // LDS byte address: low 32 bits of the flat shared pointer (ISA 10.2:
  // LDS aperture).
  unsigned int lds_addr =
      (unsigned int)(unsigned long long)(&BC[threadIdx.x & 31]);
  for (int t = 0; t < kT; ++t) {
    size_t m = (size_t)b * kT + t;
    if (threadIdx.x < 32) {   // wave 0 exactly: EXEC all-ones
      unsigned long long gaddr =
          (unsigned long long)(dbl + m * kDBL + kDTR + threadIdx.x);
      asm volatile("global_load_async_to_lds_b32 %0, %1, off"
                   :: "v"(lds_addr), "v"(gaddr) : "memory");
      asm volatile("s_wait_asynccnt 0" ::: "memory");
    }
    __syncthreads();
    float dt = delta[m * kDI + d];
    float u  = xm[m * kDI + d];
    float du = dt * u;
    float acc = 0.0f;
#pragma unroll
    for (int n = 0; n < kNS; ++n) {
      h[n] = __expf(dt * A[n]) * h[n] + du * BC[n];
      acc += h[n] * BC[kNS + n];
    }
    y[m * kDI + d] = acc + u * dparam;
    __syncthreads();
  }
}

// ------------------------------------------- gate: y * silu(z) -> bf16 ------
__global__ void gate_kernel(const float* __restrict__ y,
                            const float* __restrict__ xz,
                            __bf16* __restrict__ yg) {
  int idx = blockIdx.x * blockDim.x + threadIdx.x;
  if (idx >= kM * kDI) return;
  int d = idx % kDI;
  int m = idx / kDI;
  float z = xz[(size_t)m * kNX + kDI + d];
  float s = z / (1.0f + __expf(-z));
  yg[idx] = (__bf16)(y[idx] * s);
}

// --------------------------- final residual add + layernorm -> d_out --------
__global__ void __launch_bounds__(256)
final_norm_kernel(const float* __restrict__ hin, float* __restrict__ resid,
                  const float* __restrict__ gw, const float* __restrict__ gb,
                  float* __restrict__ out) {
  __shared__ float s1[256];
  __shared__ float s2[256];
  int blk = blockIdx.x;                 // b*L + t over kept rows only
  int b = blk / kL, t = blk % kL;
  int m = b * kT + t;
  int tid = threadIdx.x;
  size_t base = (size_t)m * kD;
  int c0 = tid, c1 = tid + 256;
  float a0 = hin[base + c0] + resid[base + c0];
  bool has1 = (c1 < kD);
  float a1 = has1 ? (hin[base + c1] + resid[base + c1]) : 0.0f;
  s1[tid] = a0 + a1;
  s2[tid] = a0 * a0 + a1 * a1;
  __syncthreads();
  for (int o = 128; o > 0; o >>= 1) {
    if (tid < o) { s1[tid] += s1[tid + o]; s2[tid] += s2[tid + o]; }
    __syncthreads();
  }
  float mean = s1[0] * (1.0f / kD);
  float var  = s2[0] * (1.0f / kD) - mean * mean;
  float inv  = rsqrtf(var + 1e-5f);
  size_t ob = (size_t)blk * kD;
  out[ob + c0] = (a0 - mean) * inv * gw[c0] + gb[c0];
  if (has1) out[ob + c1] = (a1 - mean) * inv * gw[c1] + gb[c1];
}

// ---------------------------------------------------------------------------
extern "C" void kernel_launch(void* const* d_in, const int* in_sizes, int n_in,
                              void* d_out, int out_size, void* d_ws,
                              size_t ws_size, hipStream_t stream) {
  (void)in_sizes; (void)n_in; (void)out_size; (void)ws_size;
  const int*   type_seq = (const int*)  d_in[0];
  const float* time_seq = (const float*)d_in[1];
  const float* temp_seq = (const float*)d_in[2];
  const float* emb      = (const float*)d_in[3];
  const float* norm_w   = (const float*)d_in[4];
  const float* norm_b   = (const float*)d_in[5];
  const float* in_w     = (const float*)d_in[6];
  const float* conv_w   = (const float*)d_in[7];
  const float* conv_b   = (const float*)d_in[8];
  const float* xproj_w  = (const float*)d_in[9];
  const float* dt_w     = (const float*)d_in[10];
  const float* dt_b     = (const float*)d_in[11];
  const float* A_log    = (const float*)d_in[12];
  const float* D_param  = (const float*)d_in[13];
  const float* out_w    = (const float*)d_in[14];
  const float* normf_w  = (const float*)d_in[15];
  const float* normf_b  = (const float*)d_in[16];

  char* p = (char*)d_ws;
  auto carve = [&](size_t bytes) -> void* {
    void* r = (void*)p;
    p += (bytes + 255) & ~(size_t)255;
    return r;
  };
  float*  resid = (float*) carve((size_t)kM * kD * 4);
  float*  hbuf  = (float*) carve((size_t)kM * kD * 4);
  __bf16* hnb   = (__bf16*)carve((size_t)kM * kDP * 2);
  float*  xz    = (float*) carve((size_t)kM * kNX * 4);
  float*  xm    = (float*) carve((size_t)kM * kDI * 4);
  __bf16* xmb   = (__bf16*)carve((size_t)kM * kDI * 2);
  float*  dbl   = (float*) carve((size_t)kM * kDBL * 4);
  float*  delta = (float*) carve((size_t)kM * kDI * 4);
  float*  ybuf  = (float*) carve((size_t)kM * kDI * 4);
  __bf16* ygb   = (__bf16*)carve((size_t)kM * kDI * 2);
  __bf16* winb  = (__bf16*)carve((size_t)kNX * kDP * 2);
  __bf16* woutb = (__bf16*)carve((size_t)kD * kDI * 2);

  // initial hidden state (token embed + temporal embed + temp-pad rows)
  {
    int tot = kM * kD;
    embed_kernel<<<(tot + 255) / 256, 256, 0, stream>>>(
        type_seq, time_seq, temp_seq, emb, hbuf);
  }

  for (int i = 0; i < 4; ++i) {
    const float* inw  = in_w    + (size_t)i * kNX * kD;
    const float* outw = out_w   + (size_t)i * kD * kDI;
    const float* cwi  = conv_w  + (size_t)i * kDI * 4;
    const float* cbi  = conv_b  + (size_t)i * kDI;
    const float* xwi  = xproj_w + (size_t)i * kDBL * kDI;
    const float* dtwi = dt_w    + (size_t)i * kDI * kDTR;
    const float* dtbi = dt_b    + (size_t)i * kDI;
    const float* Ali  = A_log   + (size_t)i * kDI * kNS;
    const float* Dpi  = D_param + (size_t)i * kDI;

    // stage weights as bf16 (K padded for in_proj)
    cvt_pad_kernel<<<(kNX * kDP + 255) / 256, 256, 0, stream>>>(
        inw, winb, kNX, kD, kDP);
    cvt_pad_kernel<<<(kD * kDI + 255) / 256, 256, 0, stream>>>(
        outw, woutb, kD, kDI, kDI);

    // residual add + LN -> bf16 activations (padded K)
    addnorm_kernel<<<kM, 256, 0, stream>>>(
        hbuf, resid, norm_w + i * kD, norm_b + i * kD, hnb, (i > 0) ? 1 : 0);

    // in_proj: xz = hn @ in_w^T   (8192 x 1088 x 288)
    wmma_gemm_bf16<<<dim3(kNX / 64, kM / 128), 256, 0, stream>>>(
        hnb, winb, xz, kM, kNX, kDP, kNX);

    // depthwise causal conv + SiLU
    conv_silu_kernel<<<(kM * kDI + 255) / 256, 256, 0, stream>>>(
        xz, cwi, cbi, xm, xmb);

    // x_proj (dt | B | C) and delta
    xproj_kernel<<<kM, 64, 0, stream>>>(xm, xwi, dbl);
    delta_kernel<<<(kM * kDI + 255) / 256, 256, 0, stream>>>(
        dbl, dtwi, dtbi, delta);

    // selective scan (serial in T, parallel over B*d_inner channels)
    scan_kernel<<<kB, kDI, 0, stream>>>(xm, delta, dbl, Ali, Dpi, ybuf);

    // gate with silu(z), cast to bf16
    gate_kernel<<<(kM * kDI + 255) / 256, 256, 0, stream>>>(ybuf, xz, ygb);

    // out_proj: h = (y*silu(z)) @ out_w^T   (8192 x 272 x 544)
    wmma_gemm_bf16<<<dim3((kD + 63) / 64, kM / 128), 256, 0, stream>>>(
        ygb, woutb, hbuf, kM, kD, kDI, kD);
  }

  // final residual + LN, keep only first L rows per batch
  final_norm_kernel<<<kB * kL, 256, 0, stream>>>(
      hbuf, resid, normf_w, normf_b, (float*)d_out);
}
